// TemporalFusionDecoder_24842090840197
// MI455X (gfx1250) — matrix operand order repaired and checked by reading, compiled, not run on hardware
//
#include <hip/hip_runtime.h>
#include <hip/hip_bf16.h>
#include <math.h>

#define DMODEL 512
#define S_IN   256
#define SEQL   512
#define NBATCH 64
#define LN_EPS 1e-5f
#define MBLK   64   // GEMM M-rows per block (4 stripes of 16)

typedef __attribute__((ext_vector_type(16))) __bf16 v16bf;
typedef __attribute__((ext_vector_type(8)))  float  v8f;
typedef __attribute__((ext_vector_type(4)))  unsigned int u32x4;
typedef int v4i_vs __attribute__((vector_size(4 * sizeof(int))));  // builtin's pointee type

union FragBF { v16bf v; u32x4 q[2]; };

#if defined(__has_builtin)
# if __has_builtin(__builtin_amdgcn_global_load_async_to_lds_b128)
#  define HAVE_ASYNC_LDS 1
# endif
# if __has_builtin(__builtin_amdgcn_s_wait_asynccnt)
#  define HAVE_WAIT_ASYNC 1
# endif
#endif

static __device__ inline __bf16 f2b(float f) {
  union { float f; unsigned u; } c; c.f = f;
  unsigned u = c.u;
  unsigned r = (u + 0x7FFFu + ((u >> 16) & 1u)) >> 16;
  union { unsigned short s; __bf16 b; } o; o.s = (unsigned short)r;
  return o.b;
}

static __device__ inline float sigm(float x) { return 1.f / (1.f + __expf(-x)); }

// Stage ROWS x 512 bf16 tile (arbitrary global row stride) into LDS via the
// CDNA5 async-to-LDS path (ASYNCcnt) when available.
template <int ROWS>
static __device__ inline void stage_tile(const __bf16* __restrict__ g, long rowStride,
                                         __bf16* l, int tid) {
  const int chunks = ROWS * 64;          // 16-byte chunks (512 cols * 2B / 16)
  for (int c = tid; c < chunks; c += 256) {
    int row = c >> 6;
    int col = (c & 63) * 8;
    const __bf16* gp = g + (long)row * rowStride + col;
    __bf16* lp = l + row * 512 + col;
#if defined(HAVE_ASYNC_LDS)
    __builtin_amdgcn_global_load_async_to_lds_b128((v4i_vs*)gp, (v4i_vs*)lp, 0, 0);
#else
    *(u32x4*)lp = *(const u32x4*)gp;
#endif
  }
}

static __device__ inline void wait_async_copies() {
#if defined(HAVE_ASYNC_LDS)
# if defined(HAVE_WAIT_ASYNC)
  __builtin_amdgcn_s_wait_asynccnt(0);
# else
  asm volatile("s_wait_asynccnt 0x0" ::: "memory");
# endif
#endif
}

// ---------------------------------------------------------------------------
__global__ void cast_bf16_kernel(const float* __restrict__ s, __bf16* __restrict__ d, long n) {
  long i = (long)blockIdx.x * blockDim.x + threadIdx.x;
  if (i < n) d[i] = f2b(s[i]);
}

// copy vsn_observed [B,S_IN,D] into first half of lstm_hidden [B,SEQ,D] (f32 + bf16)
__global__ void seed_hidden_kernel(const float* __restrict__ x,
                                   float* __restrict__ hidF, __bf16* __restrict__ hidB) {
  long i = (long)blockIdx.x * blockDim.x + threadIdx.x;
  long total = (long)NBATCH * S_IN * DMODEL;
  if (i >= total) return;
  int d = (int)(i % DMODEL);
  long bt = i / DMODEL;
  int t = (int)(bt % S_IN);
  int b = (int)(bt / S_IN);
  long o = ((long)b * SEQL + t) * DMODEL + d;
  float v = x[i];
  hidF[o] = v;
  hidB[o] = f2b(v);
}

// ---------------------------------------------------------------------------
// C[M,N] = A[M,K] @ B[N,K]^T (+bias1+bias2), bf16 in, f32 accum, K == 512.
// Block tile: 64(M) x 512(N); A tile staged to LDS by async copy; wave w owns
// M-stripe (w>>1) and N-half (w&1): 16 accumulators of 16x16.
// grid.x = N/512, grid.y = M/64, grid.z = batch (element strides sA/sB/sC)
__global__ __launch_bounds__(256)
void gemm_wmma_kernel(const __bf16* __restrict__ A, const __bf16* __restrict__ B,
                      const float* __restrict__ bias1, const float* __restrict__ bias2,
                      float* __restrict__ Cf, __bf16* __restrict__ Cb,
                      int N, int K, long sA, long sB, long sC) {
  __shared__ __align__(16) __bf16 aS[MBLK * 512];
  const int tid = threadIdx.x;
  const int lane = tid & 31, wave = tid >> 5;
  const int hi = lane >> 4, lo = lane & 15;
  const long z = blockIdx.z;
  A += z * sA; B += z * sB;

  stage_tile<MBLK>(A + (long)blockIdx.y * MBLK * K, K, aS, tid);
  wait_async_copies();
  __syncthreads();

  const int mi = wave >> 1;                       // 0..3
  const int col0 = blockIdx.x * 512 + (wave & 1) * 256;
  const __bf16* aRowL = aS + (mi * 16 + lo) * 512;

  v8f acc[16] = {};
  for (int k0 = 0; k0 < K; k0 += 32) {
    FragBF fa;
    fa.q[0] = *(const u32x4*)(aRowL + k0 + hi * 8);
    fa.q[1] = *(const u32x4*)(aRowL + k0 + 16 + hi * 8);
#pragma unroll
    for (int t = 0; t < 16; t++) {
      const __bf16* bRow = B + (long)(col0 + t * 16 + lo) * K + k0 + hi * 8;
      if (t == 0 && k0 + 32 < K) __builtin_prefetch(bRow + 32, 0, 1);
      FragBF fb;
      fb.q[0] = *(const u32x4*)(bRow);
      fb.q[1] = *(const u32x4*)(bRow + 16);
      acc[t] = __builtin_amdgcn_wmma_f32_16x16x32_bf16(false, fa.v, false, fb.v,
                                                       (short)0, acc[t], false, false);
    }
  }

  const int mrow0 = blockIdx.y * MBLK + mi * 16;
#pragma unroll
  for (int t = 0; t < 16; t++) {
    int n = col0 + t * 16 + lo;
    float bsum = (bias1 ? bias1[n] : 0.f) + (bias2 ? bias2[n] : 0.f);
#pragma unroll
    for (int r = 0; r < 8; r++) {
      int m = mrow0 + r + hi * 8;
      float val = acc[t][r] + bsum;
      long off = z * sC + (long)m * N + n;
      if (Cf) Cf[off] = val;
      if (Cb) Cb[off] = f2b(val);
    }
  }
}

// ---------------------------------------------------------------------------
// LSTM recurrence: one block = 16 batch rows; h (bf16) lives in LDS, gates = h@Whh^T
// by WMMA each step; Xg = x@Wih^T + bih + bhh precomputed. c state in registers.
__global__ __launch_bounds__(256)
void lstm_wmma_kernel(const float* __restrict__ Xg,        // [B, S_IN, 4D]
                      const __bf16* __restrict__ Whh,      // [4D, D] bf16
                      float* __restrict__ hidF,            // [B, SEQ, D]
                      __bf16* __restrict__ hidB) {
  __shared__ __align__(16) __bf16 hS[16 * DMODEL];
  __shared__ float gS[16 * 2048];
  const int tid = threadIdx.x;
  const int lane = tid & 31, wave = tid >> 5;
  const int hi = lane >> 4, lo = lane & 15;
  const int b0 = blockIdx.x * 16;
  const int m = tid >> 4;            // 0..15 batch row in block
  const int dbase = (tid & 15) * 32; // 32 hidden cells per thread

  float cReg[32];
#pragma unroll
  for (int j = 0; j < 32; j++) cReg[j] = 0.f;
  for (int j = tid; j < 16 * DMODEL; j += 256) hS[j] = f2b(0.f);

  for (int t = 0; t < S_IN; t++) {
    __syncthreads();
    // gS = hS @ Whh^T (16 x 2048), each wave covers 16 of 128 column tiles
    for (int tn = wave; tn < 128; tn += 8) {
      int col0 = tn * 16;
      v8f acc = {};
      const __bf16* bBase = Whh + (long)(col0 + lo) * DMODEL;
      for (int k0 = 0; k0 < DMODEL; k0 += 32) {
        FragBF fa, fb;
        fa.q[0] = *(const u32x4*)&hS[lo * DMODEL + k0 + hi * 8];
        fa.q[1] = *(const u32x4*)&hS[lo * DMODEL + k0 + 16 + hi * 8];
        fb.q[0] = *(const u32x4*)(bBase + k0 + hi * 8);
        fb.q[1] = *(const u32x4*)(bBase + k0 + 16 + hi * 8);
        acc = __builtin_amdgcn_wmma_f32_16x16x32_bf16(false, fa.v, false, fb.v,
                                                      (short)0, acc, false, false);
      }
#pragma unroll
      for (int r = 0; r < 8; r++)
        gS[(r + hi * 8) * 2048 + col0 + lo] = acc[r];
    }
    __syncthreads();
    // elementwise gate update (torch gate order i,f,g,o)
    const float* xg = Xg + ((long)(b0 + m) * S_IN + t) * 2048;
    long ho = ((long)(b0 + m) * SEQL + (S_IN + t)) * (long)DMODEL;
#pragma unroll 4
    for (int j = 0; j < 32; j++) {
      int d = dbase + j;
      float gi = gS[m * 2048 + d]        + xg[d];
      float gf = gS[m * 2048 + 512 + d]  + xg[512 + d];
      float gg = gS[m * 2048 + 1024 + d] + xg[1024 + d];
      float go = gS[m * 2048 + 1536 + d] + xg[1536 + d];
      float c = sigm(gf) * cReg[j] + sigm(gi) * tanhf(gg);
      cReg[j] = c;
      float h = sigm(go) * tanhf(c);
      hS[m * DMODEL + d] = f2b(h);
      hidF[ho + d] = h;
      hidB[ho + d] = f2b(h);
    }
  }
}

// ---------------------------------------------------------------------------
// h = sigmoid(t1)*t2 + y ; LayerNorm(h)*gamma+beta.  One block per row (D=512).
__global__ __launch_bounds__(256)
void gluln_kernel(const float* __restrict__ t1, const float* __restrict__ t2,
                  const float* __restrict__ y, const float* __restrict__ gamma,
                  const float* __restrict__ beta,
                  float* __restrict__ outF, __bf16* __restrict__ outB) {
  __shared__ float rs[256];
  __shared__ float rq[256];
  const long row = blockIdx.x;
  const int tid = threadIdx.x;
  const long base = row * (long)DMODEL;
  int i0 = tid * 2;
  float h0 = sigm(t1[base + i0])     * t2[base + i0]     + y[base + i0];
  float h1 = sigm(t1[base + i0 + 1]) * t2[base + i0 + 1] + y[base + i0 + 1];
  rs[tid] = h0 + h1;
  rq[tid] = h0 * h0 + h1 * h1;
  __syncthreads();
  for (int s = 128; s > 0; s >>= 1) {
    if (tid < s) { rs[tid] += rs[tid + s]; rq[tid] += rq[tid + s]; }
    __syncthreads();
  }
  float mu = rs[0] * (1.f / DMODEL);
  float var = rq[0] * (1.f / DMODEL) - mu * mu;
  float inv = rsqrtf(var + LN_EPS);
  float o0 = (h0 - mu) * inv * gamma[i0]     + beta[i0];
  float o1 = (h1 - mu) * inv * gamma[i0 + 1] + beta[i0 + 1];
  outF[base + i0] = o0;
  outF[base + i0 + 1] = o1;
  if (outB) { outB[base + i0] = f2b(o0); outB[base + i0 + 1] = f2b(o1); }
}

// ---------------------------------------------------------------------------
// attention scores (WMMA q@k^T) + causal mask + softmax, per (batch, 16-query tile).
// q tile staged to LDS via async copy and shared by all 8 waves.
__global__ __launch_bounds__(256)
void attn_kernel(const __bf16* __restrict__ qkv,   // [B*SEQ, 1536]
                 float* __restrict__ attnF,        // [B, SEQ, SEQ]
                 __bf16* __restrict__ attnB) {
  __shared__ __align__(16) __bf16 qS[16 * 512];
  __shared__ float sS[16 * SEQL];
  __shared__ float red[16 * 16];
  const int tid = threadIdx.x;
  const int lane = tid & 31, wave = tid >> 5;
  const int hi = lane >> 4, lo = lane & 15;
  const int b = blockIdx.y, qt = blockIdx.x;
  const float scale = 0.044194173824159216f; // 1/sqrt(512)

  const __bf16* qBase = qkv + (long)b * SEQL * 1536;
  stage_tile<16>(qBase + (long)(qt * 16) * 1536, 1536, qS, tid);
  wait_async_copies();
  __syncthreads();

  const __bf16* aRowL = qS + lo * 512;
#pragma unroll
  for (int i = 0; i < 4; i++) {
    int col0 = (wave * 4 + i) * 16;
    v8f acc = {};
    const __bf16* bRow = qBase + (long)(col0 + lo) * 1536 + 512;       // k row
    for (int k0 = 0; k0 < DMODEL; k0 += 32) {
      FragBF fa, fb;
      fa.q[0] = *(const u32x4*)(aRowL + k0 + hi * 8);
      fa.q[1] = *(const u32x4*)(aRowL + k0 + 16 + hi * 8);
      fb.q[0] = *(const u32x4*)(bRow + k0 + hi * 8);
      fb.q[1] = *(const u32x4*)(bRow + k0 + 16 + hi * 8);
      acc = __builtin_amdgcn_wmma_f32_16x16x32_bf16(false, fa.v, false, fb.v,
                                                    (short)0, acc, false, false);
    }
    int n = col0 + lo;
#pragma unroll
    for (int r = 0; r < 8; r++) {
      int mg = qt * 16 + r + hi * 8;
      float v = acc[r] * scale;
      sS[(r + hi * 8) * SEQL + n] = (n > mg) ? -1e30f : v;
    }
  }
  __syncthreads();

  const int row = tid >> 4;
  const int seg = tid & 15;
  float mx = -1e30f;
  for (int j = 0; j < 32; j++) mx = fmaxf(mx, sS[row * SEQL + seg * 32 + j]);
  red[row * 16 + seg] = mx;
  __syncthreads();
  mx = -1e30f;
  for (int j = 0; j < 16; j++) mx = fmaxf(mx, red[row * 16 + j]);
  __syncthreads();
  float sum = 0.f;
  for (int j = 0; j < 32; j++) {
    float e = __expf(sS[row * SEQL + seg * 32 + j] - mx);
    sS[row * SEQL + seg * 32 + j] = e;
    sum += e;
  }
  red[row * 16 + seg] = sum;
  __syncthreads();
  sum = 0.f;
  for (int j = 0; j < 16; j++) sum += red[row * 16 + j];
  float inv = 1.f / sum;
  long base = ((long)b * SEQL + qt * 16 + row) * (long)SEQL;
  for (int j = 0; j < 32; j++) {
    float v = sS[row * SEQL + seg * 32 + j] * inv;
    attnF[base + seg * 32 + j] = v;
    attnB[base + seg * 32 + j] = f2b(v);
  }
}

// v within qkv [B*SEQ,1536] (cols 1024..1535) -> vT [B, D, SEQ] bf16
__global__ void transpose_v_kernel(const __bf16* __restrict__ qkv, __bf16* __restrict__ vT) {
  long i = (long)blockIdx.x * blockDim.x + threadIdx.x;
  long total = (long)NBATCH * SEQL * DMODEL;
  if (i >= total) return;
  int d = (int)(i % DMODEL);
  long bs = i / DMODEL;
  int s = (int)(bs % SEQL);
  int b = (int)(bs / SEQL);
  vT[((long)b * DMODEL + d) * SEQL + s] = qkv[((long)b * SEQL + s) * 1536 + 1024 + d];
}

// ---------------------------------------------------------------------------
extern "C" void kernel_launch(void* const* d_in, const int* in_sizes, int n_in,
                              void* d_out, int out_size, void* d_ws, size_t ws_size,
                              hipStream_t stream) {
  const float* x       = (const float*)d_in[0];
  const float* Wih     = (const float*)d_in[1];
  const float* Whh     = (const float*)d_in[2];
  const float* bih     = (const float*)d_in[3];
  const float* bhh     = (const float*)d_in[4];
  const float* g1w1    = (const float*)d_in[5];
  const float* g1b1    = (const float*)d_in[6];
  const float* g1w2    = (const float*)d_in[7];
  const float* g1b2    = (const float*)d_in[8];
  const float* g1gamma = (const float*)d_in[9];
  const float* g1beta  = (const float*)d_in[10];
  const float* wqkv    = (const float*)d_in[11];
  const float* bqkv    = (const float*)d_in[12];
  const float* wo      = (const float*)d_in[13];
  const float* bo      = (const float*)d_in[14];
  const float* g2w1    = (const float*)d_in[15];
  const float* g2b1    = (const float*)d_in[16];
  const float* g2w2    = (const float*)d_in[17];
  const float* g2b2    = (const float*)d_in[18];
  const float* g2gamma = (const float*)d_in[19];
  const float* g2beta  = (const float*)d_in[20];

  float* outDelta = (float*)d_out;                                   // [B,SEQ,D]
  float* outPhi   = outDelta + (long)NBATCH * SEQL * DMODEL;         // [B,SEQ,D]
  float* outAttn  = outPhi   + (long)NBATCH * SEQL * DMODEL;         // [B,SEQ,SEQ]

  size_t off = 0;
  auto alloc = [&](size_t bytes) -> void* {
    off = (off + 255) & ~(size_t)255;
    void* p = (char*)d_ws + off;
    off += bytes;
    return p;
  };
  const long MD  = (long)NBATCH * SEQL;          // 32768 rows
  const long MDI = (long)NBATCH * S_IN;          // 16384 rows

  __bf16* WihB  = (__bf16*)alloc((size_t)2048 * 512 * 2);
  __bf16* WhhB  = (__bf16*)alloc((size_t)2048 * 512 * 2);
  __bf16* g1w1B = (__bf16*)alloc((size_t)512 * 512 * 2);
  __bf16* g1w2B = (__bf16*)alloc((size_t)512 * 512 * 2);
  __bf16* wqkvB = (__bf16*)alloc((size_t)1536 * 512 * 2);
  __bf16* woB   = (__bf16*)alloc((size_t)512 * 512 * 2);
  __bf16* g2w1B = (__bf16*)alloc((size_t)512 * 512 * 2);
  __bf16* g2w2B = (__bf16*)alloc((size_t)512 * 512 * 2);
  __bf16* xB    = (__bf16*)alloc((size_t)MDI * DMODEL * 2);
  float*  hidF  = (float*) alloc((size_t)MD * DMODEL * 4);
  __bf16* hidB  = (__bf16*)alloc((size_t)MD * DMODEL * 2);
  float*  Xg    = (float*) alloc((size_t)MDI * 2048 * 4);
  float*  t1f   = (float*) alloc((size_t)MD * DMODEL * 4);
  float*  t2f   = (float*) alloc((size_t)MD * DMODEL * 4);
  __bf16* phiB  = (__bf16*)alloc((size_t)MD * DMODEL * 2);
  __bf16* qkvB  = (__bf16*)alloc((size_t)MD * 1536 * 2);
  __bf16* vTB   = (__bf16*)alloc((size_t)MD * DMODEL * 2);
  __bf16* attnB = (__bf16*)alloc((size_t)NBATCH * SEQL * SEQL * 2);
  __bf16* ctxB  = (__bf16*)alloc((size_t)MD * DMODEL * 2);
  __bf16* ctx2B = (__bf16*)alloc((size_t)MD * DMODEL * 2);
  (void)ws_size; (void)n_in; (void)in_sizes; (void)out_size;

  auto cast = [&](const float* s, __bf16* d, long n) {
    cast_bf16_kernel<<<dim3((unsigned)((n + 255) / 256)), 256, 0, stream>>>(s, d, n);
  };
  cast(Wih,  WihB,  (long)2048 * 512);
  cast(Whh,  WhhB,  (long)2048 * 512);
  cast(g1w1, g1w1B, (long)512 * 512);
  cast(g1w2, g1w2B, (long)512 * 512);
  cast(wqkv, wqkvB, (long)1536 * 512);
  cast(wo,   woB,   (long)512 * 512);
  cast(g2w1, g2w1B, (long)512 * 512);
  cast(g2w2, g2w2B, (long)512 * 512);
  cast(x,    xB,    MDI * DMODEL);

  seed_hidden_kernel<<<dim3((unsigned)((MDI * DMODEL + 255) / 256)), 256, 0, stream>>>(x, hidF, hidB);

  // Xg = x @ Wih^T + bih + bhh   [16384 x 2048]
  gemm_wmma_kernel<<<dim3(4, (unsigned)(MDI / MBLK), 1), 256, 0, stream>>>(
      xB, WihB, bih, bhh, Xg, nullptr, 2048, 512, 0, 0, 0);

  // LSTM recurrence: fills hidF/hidB rows [S_IN, SEQ)
  lstm_wmma_kernel<<<dim3(NBATCH / 16), 256, 0, stream>>>(Xg, WhhB, hidF, hidB);

  // GLULN1
  gemm_wmma_kernel<<<dim3(1, (unsigned)(MD / MBLK), 1), 256, 0, stream>>>(
      hidB, g1w1B, g1b1, nullptr, t1f, nullptr, 512, 512, 0, 0, 0);
  gemm_wmma_kernel<<<dim3(1, (unsigned)(MD / MBLK), 1), 256, 0, stream>>>(
      hidB, g1w2B, g1b2, nullptr, t2f, nullptr, 512, 512, 0, 0, 0);
  gluln_kernel<<<dim3((unsigned)MD), 256, 0, stream>>>(
      t1f, t2f, hidF, g1gamma, g1beta, outPhi, phiB);

  // QKV projection -> bf16 [32768 x 1536]
  gemm_wmma_kernel<<<dim3(3, (unsigned)(MD / MBLK), 1), 256, 0, stream>>>(
      phiB, wqkvB, bqkv, nullptr, nullptr, qkvB, 1536, 512, 0, 0, 0);

  // masked softmax attention probabilities (writes attn output + bf16 copy)
  attn_kernel<<<dim3(SEQL / 16, NBATCH), 256, 0, stream>>>(qkvB, outAttn, attnB);

  // v^T for the attn @ v GEMM
  transpose_v_kernel<<<dim3((unsigned)((MD * DMODEL + 255) / 256)), 256, 0, stream>>>(qkvB, vTB);

  // ctx = attn @ v  (batched over 64)
  gemm_wmma_kernel<<<dim3(1, SEQL / MBLK, NBATCH), 256, 0, stream>>>(
      attnB, vTB, nullptr, nullptr, nullptr, ctxB, 512, 512,
      (long)SEQL * SEQL, (long)DMODEL * SEQL, (long)SEQL * DMODEL);

  // output projection
  gemm_wmma_kernel<<<dim3(1, (unsigned)(MD / MBLK), 1), 256, 0, stream>>>(
      ctxB, woB, bo, nullptr, nullptr, ctx2B, 512, 512, 0, 0, 0);

  // GLULN2 (residual from glu_phi) -> glu_delta
  gemm_wmma_kernel<<<dim3(1, (unsigned)(MD / MBLK), 1), 256, 0, stream>>>(
      ctx2B, g2w1B, g2b1, nullptr, t1f, nullptr, 512, 512, 0, 0, 0);
  gemm_wmma_kernel<<<dim3(1, (unsigned)(MD / MBLK), 1), 256, 0, stream>>>(
      ctx2B, g2w2B, g2b2, nullptr, t2f, nullptr, 512, 512, 0, 0, 0);
  gluln_kernel<<<dim3((unsigned)MD), 256, 0, stream>>>(
      t1f, t2f, outPhi, g2gamma, g2beta, outDelta, nullptr);
}
